// MultiHeadDotProductAttention_28372553957919
// MI455X (gfx1250) — matrix operand order repaired and checked by reading, not compiled, not run on hardware
//
#include <hip/hip_runtime.h>
#include <hip/hip_bf16.h>
#include <math.h>

#define B_  2
#define T_  2048
#define S_  2048
#define D_  1024
#define H_  8
#define M_  8
#define HD_ 128
#define BT_ (B_*T_)

typedef __bf16 bf16;
typedef __attribute__((ext_vector_type(16))) __bf16 v16bf;
typedef __attribute__((ext_vector_type(8)))  __bf16 v8bf;
typedef __attribute__((ext_vector_type(8)))  float  v8f;

__device__ __forceinline__ v8f wmma_bf16(v16bf a, v16bf b, v8f c) {
  return __builtin_amdgcn_wmma_f32_16x16x32_bf16(false, a, false, b, (short)0, c, false, false);
}

// A operand: 16x32 bf16 tile, element (row,k) at src[row*ld + k] (row-major).
// lanes0-15 row=lane, K 0..7 & 16..23; lanes16-31 row=lane-16, K 8..15 & 24..31.
__device__ __forceinline__ v16bf load_a16x32(const bf16* src, int ld, int lane) {
  int row = lane & 15;
  int k0  = (lane & 16) ? 8 : 0;
  const bf16* p = src + row * ld + k0;
  v8bf lo = *(const v8bf*)(p);
  v8bf hi = *(const v8bf*)(p + 16);
  return __builtin_shufflevector(lo, hi, 0,1,2,3,4,5,6,7,8,9,10,11,12,13,14,15);
}
// B operand: 32x16 bf16 tile, element (k,n) at src[n*ld + k] ("n-major").
// lanes0-15: col=lane, K 0..7 & 16..23; lanes16-31: col=lane-16, K 8..15 & 24..31.
__device__ __forceinline__ v16bf load_b32x16(const bf16* src, int ld, int lane) {
  int n  = lane & 15;
  int k0 = (lane & 16) ? 8 : 0;
  const bf16* p = src + n * ld + k0;
  v8bf lo = *(const v8bf*)(p);
  v8bf hi = *(const v8bf*)(p + 16);
  return __builtin_shufflevector(lo, hi, 0,1,2,3,4,5,6,7,8,9,10,11,12,13,14,15);
}

// Ping-pong GEMM helpers: prefix S names one of two operand register sets.
#define GEMM_DECL(S) v16bf S##a0, S##a1, S##b0, S##b1, S##b2, S##b3;
#define GEMM_LOAD(S, koff)                              \
  S##a0 = load_a16x32(A0 + (koff), LDA, lane);          \
  S##a1 = load_a16x32(A1 + (koff), LDA, lane);          \
  S##b0 = load_b32x16(Bp0 + (koff), LDB, lane);         \
  S##b1 = load_b32x16(Bp1 + (koff), LDB, lane);         \
  S##b2 = load_b32x16(Bp2 + (koff), LDB, lane);         \
  S##b3 = load_b32x16(Bp3 + (koff), LDB, lane);
#define GEMM_MMA(S)                                     \
  acc[0] = wmma_bf16(S##a0, S##b0, acc[0]);             \
  acc[1] = wmma_bf16(S##a1, S##b0, acc[1]);             \
  acc[2] = wmma_bf16(S##a0, S##b1, acc[2]);             \
  acc[3] = wmma_bf16(S##a1, S##b1, acc[3]);             \
  acc[4] = wmma_bf16(S##a0, S##b2, acc[4]);             \
  acc[5] = wmma_bf16(S##a1, S##b2, acc[5]);             \
  acc[6] = wmma_bf16(S##a0, S##b3, acc[6]);             \
  acc[7] = wmma_bf16(S##a1, S##b3, acc[7]);
// Branch-free pipelined k-loop: last pair peeled so the loop body has no
// conditional loads (keeps accumulator tuples stable, no v_mov/v_nop churn).
// Requires GEMM_LOAD(u, 0) already done. Handles odd #steps (causal bound).
#define GEMM_KLOOP(KDIM)                                \
  {                                                     \
    const int nsteps = (KDIM) >> 5;                     \
    const int pairs  = nsteps >> 1;                     \
    if (pairs > 0) {                                    \
      for (int it = 0; it < pairs - 1; ++it) {          \
        const int kbase = it*64;                        \
        __builtin_prefetch(A0 + kbase + 160, 0, 3);     \
        GEMM_LOAD(w, kbase + 32);                       \
        GEMM_MMA(u);                                    \
        GEMM_LOAD(u, kbase + 64);                       \
        GEMM_MMA(w);                                    \
      }                                                 \
      const int kfin = (pairs - 1)*64;                  \
      GEMM_LOAD(w, kfin + 32);                          \
      GEMM_MMA(u);                                      \
      if (nsteps & 1) {                                 \
        GEMM_LOAD(u, kfin + 64);                        \
        GEMM_MMA(w);                                    \
        GEMM_MMA(u);                                    \
      } else {                                          \
        GEMM_MMA(w);                                    \
      }                                                 \
    } else {                                            \
      GEMM_MMA(u);                                      \
    }                                                   \
  }

// ---------------- converts ----------------
__global__ void k_cvt_x(const float* __restrict__ x, bf16* __restrict__ xb) {
  int i = blockIdx.x * 256 + threadIdx.x;
  if (i < BT_*D_) xb[i] = (bf16)x[i];
}
__global__ void k_cvt_wqkv(const float* __restrict__ Wq, const float* __restrict__ Wk,
                           const float* __restrict__ Wv, bf16* __restrict__ wT) {
  int i = blockIdx.x * 256 + threadIdx.x;        // i = n*D + d, n in [0,3D)
  if (i < 3*D_*D_) {
    int n = i >> 10, d = i & (D_-1);
    const float* src = (n < D_) ? Wq : ((n < 2*D_) ? Wk : Wv);
    wT[i] = (bf16)src[d*D_ + (n & (D_-1))];
  }
}
__global__ void k_cvt_wo(const float* __restrict__ Wo, bf16* __restrict__ woT) {
  int i = blockIdx.x * 256 + threadIdx.x;        // i = e*D + d
  if (i < D_*D_) {
    int e = i >> 10, d = i & (D_-1);
    woT[i] = (bf16)Wo[d*D_ + e];
  }
}

// ---------------- per-token dynamic weights ----------------
// wdyn layout: [B,T][c=4][ii=4][m=8] f32 (ii 0..1 = w1 rms-normed, 2..3 = w2)
// ddo  layout: [B,T][c=4][m=8] f32 (tanh applied)
__global__ void k_dynw(const float* __restrict__ x, const float* __restrict__ dw1,
                       const float* __restrict__ qkw, const float* __restrict__ ddk,
                       float* __restrict__ wdyn, float* __restrict__ ddo)
{
  const int tok = blockIdx.x;                    // b*T + t
  __shared__ float xs[D_];
  __shared__ float dwh[128];                     // [4][32]
  for (int i = threadIdx.x; i < D_; i += 128) xs[i] = x[tok*D_ + i];
  __syncthreads();
  const int tid = threadIdx.x;
  float acc = 0.f;
  for (int d = 0; d < D_; ++d) acc = fmaf(xs[d], dw1[d*128 + tid], acc);
  float u = acc;                                  // gelu (tanh approx, jax default)
  dwh[tid] = 0.5f*u*(1.f + tanhf(0.7978845608028654f*(u + 0.044715f*u*u*u)));
  __syncthreads();
  const int c = tid >> 5, rest = tid & 31;        // rest = ii*8 + m
  float w = 0.f;
  for (int k = 0; k < 32; ++k)
    w = fmaf(dwh[c*32 + k], qkw[((c*32 + k)*4 + (rest >> 3))*8 + (rest & 7)], w);
  float ss = w*w;
  ss += __shfl_xor(ss, 1); ss += __shfl_xor(ss, 2); ss += __shfl_xor(ss, 4);
  if ((rest >> 3) < 2) w *= rsqrtf(ss*0.125f + 1e-6f);   // rmsnorm over m for w1 half
  wdyn[tok*128 + tid] = w;
  if (tid < 32) {
    float a2 = 0.f;
    for (int d = 0; d < D_; ++d) a2 = fmaf(xs[d], ddk[d*32 + tid], a2);
    ddo[tok*32 + tid] = tanhf(a2);
  }
}

// ---------------- QKV GEMM (bf16 WMMA, 32x64 wave tile, ping-pong pipelined) -----
// grid: (BT/32, 3D/256), block 128 (4 waves, each 32x64 tile)
__global__ void __launch_bounds__(128, 1)
k_qkv_gemm(const bf16* __restrict__ xin, const bf16* __restrict__ wT,
           const float* __restrict__ bq, const float* __restrict__ bk,
           const float* __restrict__ bv,
           bf16* __restrict__ qb, bf16* __restrict__ kb, bf16* __restrict__ vT)
{
  const int lane = threadIdx.x & 31;
  const int wave = threadIdx.x >> 5;
  const int r0 = blockIdx.x * 32;
  const int n0 = blockIdx.y * 256 + wave * 64;
  const int LDA = D_, LDB = D_;
  const bf16* A0  = xin + r0*D_;
  const bf16* A1  = xin + (r0 + 16)*D_;
  const bf16* Bp0 = wT + (n0 +  0)*D_;
  const bf16* Bp1 = wT + (n0 + 16)*D_;
  const bf16* Bp2 = wT + (n0 + 32)*D_;
  const bf16* Bp3 = wT + (n0 + 48)*D_;
  v8f acc[8] = {};
  GEMM_DECL(u) GEMM_DECL(w)
  GEMM_LOAD(u, 0);
  GEMM_KLOOP(D_);

  const float qscale = 0.08838834764831845f;     // 1/sqrt(128)
  const int hi8 = (lane & 16) ? 8 : 0;
  const int nn = lane & 15;
  #pragma unroll
  for (int j = 0; j < 4; ++j) {
    const int n = n0 + j*16 + nn;
    const int seg = n >> 10;                     // 0=q,1=k,2=v
    const int cidx = n & (D_-1);
    const int h = cidx >> 7, hd = cidx & 127;
    const float bias = (seg == 0 ? bq : (seg == 1 ? bk : bv))[cidx];
    #pragma unroll
    for (int rt = 0; rt < 2; ++rt) {
      #pragma unroll
      for (int e = 0; e < 8; ++e) {
        const int r = r0 + rt*16 + e + hi8;
        const int b = r >> 11;
        const int t = r & (T_-1);
        const float v = acc[j*2 + rt][e] + bias;
        if (seg == 0)      qb[((b*H_ + h)*T_ + t)*HD_ + hd] = (bf16)(v * qscale);
        else if (seg == 1) kb[((b*H_ + h)*T_ + t)*HD_ + hd] = (bf16)v;
        else               vT[((b*H_ + h)*HD_ + hd)*S_ + t] = (bf16)v;   // V transposed
      }
    }
  }
}

// ---------------- logits = QK^T (all 8 heads) + pre cross-head mix + causal mask ----
// grid: B*(T/16), block 128 (4 waves; wave handles every 4th s-tile). bf16 out.
__global__ void __launch_bounds__(128, 1)
k_logits(const bf16* __restrict__ qb, const bf16* __restrict__ kb,
         const float* __restrict__ wdyn, const float* __restrict__ ddo,
         const float* __restrict__ prw, bf16* __restrict__ lg)
{
  const int t16 = blockIdx.x & (T_/16 - 1);
  const int b   = blockIdx.x >> 7;
  const int t0  = t16 * 16;

  __shared__ bf16  qs[M_][16][HD_];              // 32KB: Q tile for all heads
  __shared__ float qw_s[16][4][8];               // pre_q w1/w2 per t
  __shared__ float qdd_s[16][8];
  __shared__ float prew_s[8][8];

  for (int i = threadIdx.x; i < M_*16*(HD_/8); i += 128) {
    int m   = i / (16*(HD_/8));
    int rem = i - m*(16*(HD_/8));
    int row = rem / (HD_/8);
    int kc  = rem - row*(HD_/8);
    *(((uint4*)&qs[m][row][0]) + kc) =
        *(const uint4*)(qb + ((b*H_ + m)*T_ + (t0 + row))*HD_ + kc*8);
  }
  for (int i = threadIdx.x; i < 16*32; i += 128) {
    int row = i >> 5, q = i & 31;
    qw_s[row][q >> 3][q & 7] = wdyn[(b*T_ + t0 + row)*128 + q];       // c=0
  }
  for (int i = threadIdx.x; i < 16*8; i += 128)
    qdd_s[i >> 3][i & 7] = ddo[(b*T_ + t0 + (i >> 3))*32 + (i & 7)];  // c=0
  if (threadIdx.x < 64) prew_s[threadIdx.x >> 3][threadIdx.x & 7] = prw[threadIdx.x];
  __syncthreads();

  const int lane = threadIdx.x & 31;
  const int wave = threadIdx.x >> 5;
  const int ntiles = (t16 + 2) & ~1;             // pad to 32-wide K for PV GEMM

  for (int st = wave; st < ntiles; st += 4) {
    const int s0 = st * 16;
    v8f acc[8] = {};
    if (s0 <= t0 + 15) {                         // wave-uniform: skip WMMA on fully-masked tile
      const bf16* Kbase = kb + (b*H_*T_ + s0)*HD_;   // + m*T_*HD_ + kc*32
      v16bf bcur = load_b32x16(Kbase, HD_, lane);
      #pragma unroll
      for (int idx = 0; idx < 32; ++idx) {       // idx = m*4 + kc, B preloaded 1 deep
        const int m = idx >> 2, kc = idx & 3;
        v16bf bnext = bcur;
        if (idx < 31) {
          const int m2 = (idx + 1) >> 2, kc2 = (idx + 1) & 3;
          bnext = load_b32x16(Kbase + m2*(T_*HD_) + kc2*32, HD_, lane);
        }
        v16bf a = load_a16x32(&qs[m][0][0] + kc*32, HD_, lane);
        acc[m] = wmma_bf16(a, bcur, acc[m]);
        bcur = bnext;
      }
    }
    const int s = s0 + (lane & 15);
    const float* kwp = wdyn + (b*T_ + s)*128 + 32;                    // c=1 (pre_k)
    float kw1r[2][8], kw2r[2][8], kddr[8];
    #pragma unroll
    for (int i2 = 0; i2 < 2; ++i2)
      #pragma unroll
      for (int m = 0; m < 8; ++m) { kw1r[i2][m] = kwp[i2*8 + m]; kw2r[i2][m] = kwp[16 + i2*8 + m]; }
    #pragma unroll
    for (int m = 0; m < 8; ++m) kddr[m] = ddo[(b*T_ + s)*32 + 8 + m];

    const int hi8 = (lane & 16) ? 8 : 0;
    for (int j = 0; j < 8; ++j) {
      const int tl = j + hi8;
      const int tg = t0 + tl;
      float inp[8], ret[8];
      #pragma unroll
      for (int m = 0; m < 8; ++m) inp[m] = acc[m][j];
      #pragma unroll
      for (int n = 0; n < 8; ++n) {
        float r = inp[n];
        #pragma unroll
        for (int m = 0; m < 8; ++m) r = fmaf(inp[m], prew_s[m][n], r);
        ret[n] = r;
      }
      #pragma unroll
      for (int i2 = 0; i2 < 2; ++i2) {
        float hq = 0.f, hk = 0.f;
        #pragma unroll
        for (int m = 0; m < 8; ++m) {
          hq = fmaf(inp[m], qw_s[tl][i2][m], hq);
          hk = fmaf(inp[m], kw1r[i2][m], hk);
        }
        #pragma unroll
        for (int n = 0; n < 8; ++n) {
          ret[n] = fmaf(hq, qw_s[tl][2 + i2][n], ret[n]);
          ret[n] = fmaf(hk, kw2r[i2][n], ret[n]);
        }
      }
      #pragma unroll
      for (int n = 0; n < 8; ++n) ret[n] += inp[n]*qdd_s[tl][n] + inp[n]*kddr[n];
      const bool msk = (s > tg);
      #pragma unroll
      for (int n = 0; n < 8; ++n) {
        float v = msk ? -1.0e30f : ret[n];
        lg[((b*M_ + n)*T_ + tg)*S_ + s] = (bf16)v;
      }
    }
  }
}

// ---------------- row softmax stats (online max+sum) ----------------
__global__ void k_stats(const bf16* __restrict__ lg, float* __restrict__ stats)
{
  const int row = blockIdx.x;                    // (b*M + m)*T + t
  const int t = row & (T_-1);
  const int bound = (((t >> 4) + 2) & ~1) << 4;
  const bf16* L = lg + (size_t)row * S_;
  float mu = -3.0e38f, z = 0.f;
  for (int s = threadIdx.x; s < bound; s += 128) {
    const float l = (float)L[s];
    const float Mx = fmaxf(mu, l);
    z = z * __expf(mu - Mx) + __expf(l - Mx);
    mu = Mx;
  }
  for (int o = 16; o > 0; o >>= 1) {
    const float mu2 = __shfl_xor(mu, o), z2 = __shfl_xor(z, o);
    const float Mx = fmaxf(mu, mu2);
    z = z * __expf(mu - Mx) + z2 * __expf(mu2 - Mx);
    mu = Mx;
  }
  __shared__ float smu[4], szz[4];
  const int lane = threadIdx.x & 31, wv = threadIdx.x >> 5;
  if (lane == 0) { smu[wv] = mu; szz[wv] = z; }
  __syncthreads();
  if (threadIdx.x == 0) {
    mu = smu[0]; z = szz[0];
    for (int i = 1; i < 4; ++i) {
      const float Mx = fmaxf(mu, smu[i]);
      z = z * __expf(mu - Mx) + szz[i] * __expf(smu[i] - Mx);
      mu = Mx;
    }
    stats[row*2] = mu; stats[row*2 + 1] = z;
  }
}

// ---------------- probs = softmax + post cross-head mix, IN PLACE on lg ----------
__global__ void k_probs(bf16* __restrict__ lg, const float* __restrict__ stats,
                        const float* __restrict__ wdyn, const float* __restrict__ ddo,
                        const float* __restrict__ pw)
{
  const int t16 = blockIdx.x & (T_/16 - 1);
  const int b   = blockIdx.x >> 7;
  const int t0  = t16 * 16;
  __shared__ float mu_s[8][16], iz_s[8][16];
  __shared__ float qw_s[16][4][8];
  __shared__ float qdd_s[16][8];
  __shared__ float pw_s[8][8];
  for (int i = threadIdx.x; i < 128; i += 256) {
    const int m = i >> 4, r = i & 15;
    const int row = (b*M_ + m)*T_ + t0 + r;
    mu_s[m][r] = stats[row*2];
    iz_s[m][r] = 1.f / stats[row*2 + 1];
  }
  for (int i = threadIdx.x; i < 16*32; i += 256) {
    const int r = i >> 5, q = i & 31;
    qw_s[r][q >> 3][q & 7] = wdyn[(b*T_ + t0 + r)*128 + 64 + q];      // c=2 (post_q)
  }
  for (int i = threadIdx.x; i < 16*8; i += 256)
    qdd_s[i >> 3][i & 7] = ddo[(b*T_ + t0 + (i >> 3))*32 + 16 + (i & 7)];
  if (threadIdx.x < 64) pw_s[threadIdx.x >> 3][threadIdx.x & 7] = pw[threadIdx.x];
  __syncthreads();

  const int bound = ((t16 + 2) & ~1) * 16;
  for (int s = threadIdx.x; s < bound; s += 256) {
    const float* kwp = wdyn + (b*T_ + s)*128 + 96;                    // c=3 (post_k)
    float kw1r[2][8], kw2r[2][8], kddr[8];
    #pragma unroll
    for (int i2 = 0; i2 < 2; ++i2)
      #pragma unroll
      for (int m = 0; m < 8; ++m) { kw1r[i2][m] = kwp[i2*8 + m]; kw2r[i2][m] = kwp[16 + i2*8 + m]; }
    #pragma unroll
    for (int m = 0; m < 8; ++m) kddr[m] = ddo[(b*T_ + s)*32 + 24 + m];

    for (int r = 0; r < 16; ++r) {
      const int tg = t0 + r;
      float p[8], ret[8];
      #pragma unroll
      for (int m = 0; m < 8; ++m) {
        const float l = (float)lg[((b*M_ + m)*T_ + tg)*S_ + s];
        p[m] = __expf(l - mu_s[m][r]) * iz_s[m][r];                   // masked -> 0
      }
      #pragma unroll
      for (int n = 0; n < 8; ++n) {
        float rr = p[n];
        #pragma unroll
        for (int m = 0; m < 8; ++m) rr = fmaf(p[m], pw_s[m][n], rr);
        ret[n] = rr;
      }
      #pragma unroll
      for (int i2 = 0; i2 < 2; ++i2) {
        float hq = 0.f, hk = 0.f;
        #pragma unroll
        for (int m = 0; m < 8; ++m) { hq = fmaf(p[m], qw_s[r][i2][m], hq); hk = fmaf(p[m], kw1r[i2][m], hk); }
        #pragma unroll
        for (int n = 0; n < 8; ++n) { ret[n] = fmaf(hq, qw_s[r][2 + i2][n], ret[n]); ret[n] = fmaf(hk, kw2r[i2][n], ret[n]); }
      }
      #pragma unroll
      for (int n = 0; n < 8; ++n) ret[n] += p[n]*qdd_s[r][n] + p[n]*kddr[n];
      #pragma unroll
      for (int n = 0; n < 8; ++n) lg[((b*M_ + n)*T_ + tg)*S_ + s] = (bf16)ret[n];
    }
  }
}

// ---------------- PV GEMM: out = probs_mixed x V (WMMA, 32x64, ping-pong) --------
// block 128 = 4 waves; wave job = (b, m, t32, hd-half of 64)
__global__ void __launch_bounds__(128, 1)
k_pv(const bf16* __restrict__ pm, const bf16* __restrict__ vT,
     bf16* __restrict__ attnb)
{
  const int lane = threadIdx.x & 31;
  int job = blockIdx.x * 4 + (threadIdx.x >> 5);
  const int half = job & 1; job >>= 1;
  const int t32 = job & (T_/32 - 1); job >>= 6;
  const int m = job & 7; job >>= 3;
  const int b = job;
  const int t0 = t32 * 32;
  const int kbound = (t32 + 1) * 32;             // causal bound, mult of 32; pad region = zero probs
  const int LDA = S_, LDB = S_;
  const bf16* A0  = pm + ((b*M_ + m)*T_ + t0)*S_;
  const bf16* A1  = A0 + 16*S_;
  const bf16* Bp0 = vT + ((b*H_ + m)*HD_ + half*64)*S_;
  const bf16* Bp1 = Bp0 + 16*S_;
  const bf16* Bp2 = Bp0 + 32*S_;
  const bf16* Bp3 = Bp0 + 48*S_;
  v8f acc[8] = {};
  GEMM_DECL(u) GEMM_DECL(w)
  GEMM_LOAD(u, 0);
  GEMM_KLOOP(kbound);

  const int hi8 = (lane & 16) ? 8 : 0;
  const int nn = lane & 15;
  #pragma unroll
  for (int j = 0; j < 4; ++j) {
    const int col = m*HD_ + half*64 + j*16 + nn;
    #pragma unroll
    for (int rt = 0; rt < 2; ++rt) {
      #pragma unroll
      for (int e = 0; e < 8; ++e) {
        const int t = t0 + rt*16 + e + hi8;
        attnb[(b*T_ + t)*D_ + col] = (bf16)acc[j*2 + rt][e];
      }
    }
  }
}

// ---------------- output projection GEMM (32x64, ping-pong) ----------------
__global__ void __launch_bounds__(128, 1)
k_out_gemm(const bf16* __restrict__ attnb, const bf16* __restrict__ woT,
           const float* __restrict__ bo, float* __restrict__ out)
{
  const int lane = threadIdx.x & 31;
  const int wave = threadIdx.x >> 5;
  const int r0 = blockIdx.x * 32;
  const int n0 = blockIdx.y * 256 + wave * 64;
  const int LDA = D_, LDB = D_;
  const bf16* A0  = attnb + r0*D_;
  const bf16* A1  = attnb + (r0 + 16)*D_;
  const bf16* Bp0 = woT + (n0 +  0)*D_;
  const bf16* Bp1 = woT + (n0 + 16)*D_;
  const bf16* Bp2 = woT + (n0 + 32)*D_;
  const bf16* Bp3 = woT + (n0 + 48)*D_;
  v8f acc[8] = {};
  GEMM_DECL(u) GEMM_DECL(w)
  GEMM_LOAD(u, 0);
  GEMM_KLOOP(D_);

  const int hi8 = (lane & 16) ? 8 : 0;
  const int nn = lane & 15;
  #pragma unroll
  for (int j = 0; j < 4; ++j) {
    const int n = n0 + j*16 + nn;
    const float bias = bo[n];
    #pragma unroll
    for (int rt = 0; rt < 2; ++rt) {
      #pragma unroll
      for (int e = 0; e < 8; ++e) {
        const int r = r0 + rt*16 + e + hi8;
        out[(size_t)r*D_ + n] = acc[j*2 + rt][e] + bias;
      }
    }
  }
}

extern "C" void kernel_launch(void* const* d_in, const int* in_sizes, int n_in,
                              void* d_out, int out_size, void* d_ws, size_t ws_size,
                              hipStream_t stream)
{
  (void)in_sizes; (void)n_in; (void)out_size; (void)ws_size;
  const float* x   = (const float*)d_in[0];
  const float* Wq  = (const float*)d_in[1];
  const float* bq  = (const float*)d_in[2];
  const float* Wk  = (const float*)d_in[3];
  const float* bk  = (const float*)d_in[4];
  const float* Wv  = (const float*)d_in[5];
  const float* bv  = (const float*)d_in[6];
  const float* Wo  = (const float*)d_in[7];
  const float* bo  = (const float*)d_in[8];
  const float* dw1 = (const float*)d_in[9];
  const float* qkw = (const float*)d_in[10];
  const float* ddk = (const float*)d_in[11];
  const float* prw = (const float*)d_in[12];
  const float* pow_= (const float*)d_in[13];
  float* out = (float*)d_out;

  char* wsb = (char*)d_ws;
  size_t off = 0;
  auto take = [&](size_t bytes) -> char* {
    char* p = wsb + off;
    off = (off + bytes + 255) & ~(size_t)255;
    return p;
  };
  bf16*  xb    = (bf16*) take((size_t)BT_*D_*2);
  bf16*  wqkvT = (bf16*) take((size_t)3*D_*D_*2);
  bf16*  woT   = (bf16*) take((size_t)D_*D_*2);
  bf16*  qb    = (bf16*) take((size_t)B_*H_*T_*HD_*2);
  bf16*  kb    = (bf16*) take((size_t)B_*H_*T_*HD_*2);
  bf16*  vT    = (bf16*) take((size_t)B_*H_*HD_*S_*2);
  bf16*  attnb = (bf16*) take((size_t)BT_*D_*2);
  float* wdyn  = (float*)take((size_t)BT_*128*4);
  float* ddo   = (float*)take((size_t)BT_*32*4);
  float* stats = (float*)take((size_t)B_*M_*T_*2*4);
  bf16*  lg    = (bf16*) take((size_t)B_*M_*T_*S_*2);

  k_cvt_x   <<<(BT_*D_)/256, 256, 0, stream>>>(x, xb);
  k_cvt_wqkv<<<(3*D_*D_)/256, 256, 0, stream>>>(Wq, Wk, Wv, wqkvT);
  k_cvt_wo  <<<(D_*D_)/256, 256, 0, stream>>>(Wo, woT);
  k_dynw    <<<BT_, 128, 0, stream>>>(x, dw1, qkw, ddk, wdyn, ddo);
  k_qkv_gemm<<<dim3(BT_/32, (3*D_)/256), 128, 0, stream>>>(xb, wqkvT, bq, bk, bv, qb, kb, vT);
  k_logits  <<<B_*(T_/16), 128, 0, stream>>>(qb, kb, wdyn, ddo, prw, lg);
  k_stats   <<<B_*M_*T_, 128, 0, stream>>>(lg, stats);
  k_probs   <<<B_*(T_/16), 256, 0, stream>>>(lg, stats, wdyn, ddo, pow_);
  k_pv      <<<(B_*M_*(T_/32)*2)/4, 128, 0, stream>>>(lg, vT, attnb);
  k_out_gemm<<<dim3(BT_/32, D_/256), 128, 0, stream>>>(attnb, woT, bo, out);
}
